// MockModel_74053826118133
// MI455X (gfx1250) — matrix operand order repaired and checked by reading, not compiled, not run on hardware
//
#include <hip/hip_runtime.h>
#include <math.h>

// ---------------------------------------------------------------------------
// Hyperbolic (Poincare-ball) attention for MI455X / gfx1250 (wave32, WMMA).
// Fused flash-attention avoids the 256MB score tensor entirely; all GEMMs use
// v_wmma_f32_16x16x32_f16 (dh=32 -> exactly one WMMA per 16x16 score tile).
// K/V tiles stream into LDS with GLOBAL_LOAD_ASYNC_TO_LDS_B32 (ASYNCcnt path).
// ---------------------------------------------------------------------------

typedef __attribute__((ext_vector_type(16))) _Float16 v16h;
typedef __attribute__((ext_vector_type(8)))  float    v8f;

#define BATCH      2
#define SEQ        2048
#define D_MODEL    256
#define NUM_HEADS  8
#define DH         32
#define CURV       1.0f
#define EPS_REF    1e-5f

enum { MODE_F32_OUT = 0, MODE_F16_OUT = 1, MODE_EXPMAP_HEADS = 2, MODE_HEADS_F16 = 3 };

// 16-bit A-matrix 16x32: VGPR pair j -> K base; halves interleave at 8.
__device__ __forceinline__ int kmapA(int j, int half) {
    return (j < 4 ? 2 * j : 16 + 2 * (j - 4)) + 8 * half;
}
// 16-bit B-matrix 32x16: same K order as A (VGPR pair j -> K base; halves interleave at 8).
__device__ __forceinline__ int kmapB(int j, int half) {
    return (j < 4 ? 2 * j : 16 + 2 * (j - 4)) + 8 * half;
}

// Async global->LDS dword copy (no VGPR round-trip, tracked by ASYNCcnt).
__device__ __forceinline__ void async_copy_b32(uint32_t lds_off, const void* gptr) {
    asm volatile("global_load_async_to_lds_b32 %0, %1, off"
                 :: "v"(lds_off), "v"(gptr) : "memory");
}
__device__ __forceinline__ void wait_asynccnt0() {
    asm volatile("s_wait_asynccnt 0x0" ::: "memory");
}

// ---------------------------------------------------------------------------
// C[M,256] = A[M,256] @ W[256,256]^T + bias, with mode-dependent epilogue.
// Block: 256 threads = 8 waves, tile 64(M) x 256(N); wave = 16(M) x 128(N).
// ---------------------------------------------------------------------------
template <typename Tin>
__global__ __launch_bounds__(256)
void gemm_wt_kernel(const Tin* __restrict__ Ag, const float* __restrict__ Wg,
                    const float* __restrict__ bias, void* __restrict__ outp,
                    float* __restrict__ norms, int M, int K, int mode)
{
    __shared__ _Float16 As[64 * 34];
    __shared__ _Float16 Ws[256 * 34];

    const int tid  = threadIdx.x;
    const int wave = tid >> 5;
    const int lane = tid & 31;
    const int r    = lane & 15;
    const int half = lane >> 4;
    const int wm   = wave & 3;   // M strip (16 rows each)
    const int wn   = wave >> 2;  // N half (128 cols each)
    const int row0 = blockIdx.x * 64;

    v8f zero = {};
    v8f acc[8];
#pragma unroll
    for (int nf = 0; nf < 8; ++nf) acc[nf] = zero;

    for (int k0 = 0; k0 < K; k0 += 32) {
        for (int idx = tid; idx < 64 * 32; idx += 256) {
            int rr = idx >> 5, cc = idx & 31;
            As[rr * 34 + cc] = (_Float16)(float)Ag[(size_t)(row0 + rr) * K + k0 + cc];
        }
        for (int idx = tid; idx < 256 * 32; idx += 256) {
            int rr = idx >> 5, cc = idx & 31;
            Ws[rr * 34 + cc] = (_Float16)Wg[(size_t)rr * K + k0 + cc];
        }
        __syncthreads();

        v16h af;
#pragma unroll
        for (int j = 0; j < 8; ++j) {
            int kb = kmapA(j, half);
            const _Float16* ap = &As[(wm * 16 + r) * 34 + kb];
            af[2 * j] = ap[0]; af[2 * j + 1] = ap[1];
        }
#pragma unroll
        for (int nf = 0; nf < 8; ++nf) {
            v16h bf;
            int n_loc = wn * 128 + nf * 16 + r;
#pragma unroll
            for (int j = 0; j < 8; ++j) {
                int kb = kmapB(j, half);
                const _Float16* bp = &Ws[n_loc * 34 + kb];
                bf[2 * j] = bp[0]; bf[2 * j + 1] = bp[1];
            }
            acc[nf] = __builtin_amdgcn_wmma_f32_16x16x32_f16(
                false, af, false, bf, (short)0, acc[nf], false, false);
        }
        __syncthreads();
    }

    // bias
#pragma unroll
    for (int nf = 0; nf < 8; ++nf) {
        float bval = bias[wn * 128 + nf * 16 + r];
#pragma unroll
        for (int i = 0; i < 8; ++i) acc[nf][i] += bval;
    }

    if (mode == MODE_F32_OUT || mode == MODE_F16_OUT) {
#pragma unroll
        for (int nf = 0; nf < 8; ++nf) {
            int n = wn * 128 + nf * 16 + r;
#pragma unroll
            for (int i = 0; i < 8; ++i) {
                int m = row0 + wm * 16 + half * 8 + i;
                if (mode == MODE_F32_OUT)
                    ((float*)outp)[(size_t)m * 256 + n] = acc[nf][i];
                else
                    ((_Float16*)outp)[(size_t)m * 256 + n] = (_Float16)acc[nf][i];
            }
        }
    } else {
        // heads layout [B,H,S,DH]; optional expmap0 per 32-wide head segment
#pragma unroll
        for (int seg = 0; seg < 4; ++seg) {
            float qq8[8];
            if (mode == MODE_EXPMAP_HEADS) {
#pragma unroll
                for (int i = 0; i < 8; ++i) {
                    float s = acc[2 * seg][i] * acc[2 * seg][i]
                            + acc[2 * seg + 1][i] * acc[2 * seg + 1][i];
#pragma unroll
                    for (int off = 8; off >= 1; off >>= 1)
                        s += __shfl_xor(s, off, 16);
                    float nrm = sqrtf(fmaxf(s, 1e-12f));
                    float t   = tanhf(nrm);     // sqrt(c)=1
                    float scl = t / nrm;
                    acc[2 * seg][i]     *= scl;
                    acc[2 * seg + 1][i] *= scl;
                    qq8[i] = t * t;             // ||q||^2 after expmap0
                }
            }
            int h = wn * 4 + seg;
#pragma unroll
            for (int f = 0; f < 2; ++f) {
                int nf = 2 * seg + f;
                int d  = f * 16 + r;
#pragma unroll
                for (int i = 0; i < 8; ++i) {
                    int m  = row0 + wm * 16 + half * 8 + i;
                    int b  = m >> 11;          // S = 2048
                    int s_ = m & 2047;
                    ((_Float16*)outp)[(((size_t)(b * NUM_HEADS + h) * SEQ) + s_) * DH + d]
                        = (_Float16)acc[nf][i];
                }
            }
            if (mode == MODE_EXPMAP_HEADS && r == 0) {
#pragma unroll
                for (int i = 0; i < 8; ++i) {
                    int m  = row0 + wm * 16 + half * 8 + i;
                    int b  = m >> 11;
                    int s_ = m & 2047;
                    norms[((size_t)(b * NUM_HEADS + h) * SEQ) + s_] = qq8[i];
                }
            }
        }
    }
}

// ---------------------------------------------------------------------------
// Fused flash attention with hyperbolic-distance scores.
// Grid (S/64, H, B), 128 threads = 4 waves; each wave owns a 16-row Q strip.
// K/V tiles arrive via GLOBAL_LOAD_ASYNC_TO_LDS_B32.
// ---------------------------------------------------------------------------
__global__ __launch_bounds__(128)
void hyp_attn_kernel(const _Float16* __restrict__ Qh,   // [B,H,S,DH]
                     const _Float16* __restrict__ Kh,
                     const _Float16* __restrict__ Vh,
                     const float* __restrict__ QQ,      // [B,H,S] tanh^2 norms
                     const float* __restrict__ KK,
                     _Float16* __restrict__ AO)         // [B,S,D] f16
{
    __shared__ _Float16 Kt[64 * 34];
    __shared__ _Float16 Vt[64 * 34];
    __shared__ float    kkS[64];
    __shared__ _Float16 Pb[4][16 * 68];  // per-wave 16x64 P staging

    const int tid  = threadIdx.x;
    const int wave = tid >> 5;
    const int lane = tid & 31;
    const int r    = lane & 15;
    const int half = lane >> 4;
    const int h    = blockIdx.y;
    const int b    = blockIdx.z;
    const size_t bh = (size_t)(b * NUM_HEADS + h);

    const _Float16* Qbh = Qh + bh * SEQ * DH;
    const _Float16* Kbh = Kh + bh * SEQ * DH;
    const _Float16* Vbh = Vh + bh * SEQ * DH;
    const float* qqbh = QQ + bh * SEQ;
    const float* kkbh = KK + bh * SEQ;

    // LDS byte offsets (flat shared pointer truncates to LDS offset)
    const uint32_t ktb = (uint32_t)(uintptr_t)(&Kt[0]);
    const uint32_t vtb = (uint32_t)(uintptr_t)(&Vt[0]);
    const uint32_t kkb = (uint32_t)(uintptr_t)(&kkS[0]);

    const int q0 = blockIdx.x * 64 + wave * 16;

    // persistent Q A-fragment (dh = 32 = one WMMA K dimension)
    v16h qa;
#pragma unroll
    for (int j = 0; j < 8; ++j) {
        int kb = kmapA(j, half);
        const _Float16* qp = &Qbh[(size_t)(q0 + r) * DH + kb];
        qa[2 * j] = qp[0]; qa[2 * j + 1] = qp[1];
    }
    float qq8[8];
#pragma unroll
    for (int i = 0; i < 8; ++i) qq8[i] = qqbh[q0 + half * 8 + i];

    float m8[8], l8[8];
#pragma unroll
    for (int i = 0; i < 8; ++i) { m8[i] = -__builtin_inff(); l8[i] = 0.f; }
    v8f zero = {};
    v8f ofr[2]; ofr[0] = zero; ofr[1] = zero;

    const float inv_sqrt_dh = 0.17677669529663687f;  // 1/sqrt(32)
    const float one_m_eps   = 1.0f - EPS_REF;

    for (int kt = 0; kt < SEQ / 64; ++kt) {
        const int krow0 = kt * 64;
        // async K/V tile copy: global -> LDS, no VGPR round-trip (ASYNCcnt)
#pragma unroll
        for (int it = 0; it < 8; ++it) {
            int idx = tid + it * 128;          // 1024 dwords per 64x32 f16 tile
            int rr  = idx >> 4;
            int dc  = (idx & 15) * 2;          // f16 column (even)
            uint32_t loff = (uint32_t)(rr * 34 + dc) * 2u;
            async_copy_b32(ktb + loff, &Kbh[(size_t)(krow0 + rr) * DH + dc]);
            async_copy_b32(vtb + loff, &Vbh[(size_t)(krow0 + rr) * DH + dc]);
        }
        if (tid < 64)
            async_copy_b32(kkb + (uint32_t)tid * 4u, &kkbh[krow0 + tid]);
        if (kt + 1 < SEQ / 64) {   // global_prefetch_b8 for next tile
            __builtin_prefetch(&Kbh[(size_t)(krow0 + 64) * DH], 0, 1);
            __builtin_prefetch(&Vbh[(size_t)(krow0 + 64) * DH], 0, 1);
        }
        wait_asynccnt0();          // drain this wave's async copies
        __syncthreads();           // make whole tile visible block-wide

        // S-tile = Q K^T, then hyperbolic-distance scores in registers
        float sc[4][8];
#pragma unroll
        for (int nf = 0; nf < 4; ++nf) {
            v16h kf;
#pragma unroll
            for (int j = 0; j < 8; ++j) {
                int kb = kmapB(j, half);            // over dh
                const _Float16* kp = &Kt[(nf * 16 + r) * 34 + kb];
                kf[2 * j] = kp[0]; kf[2 * j + 1] = kp[1];
            }
            v8f sf = zero;
            sf = __builtin_amdgcn_wmma_f32_16x16x32_f16(
                false, qa, false, kf, (short)0, sf, false, false);
            float kk = kkS[nf * 16 + r];
#pragma unroll
            for (int i = 0; i < 8; ++i) {
                float qk  = sf[i];
                float qq  = qq8[i];
                float Aa  = 1.f - 2.f * CURV * qk + CURV * kk;
                float Bb  = 1.f - CURV * qq;
                float den = 1.f - 2.f * CURV * qk + CURV * CURV * qq * kk;
                float n2  = Aa * Aa * qq + Bb * Bb * kk - 2.f * Aa * Bb * qk;
                float nn  = sqrtf(fmaxf(n2, 0.f)) / fmaxf(fabsf(den), 1e-12f);
                float xx  = fminf(nn, one_m_eps);
                float dist = __logf((1.f + xx) / (1.f - xx));  // 2*atanh(x)
                sc[nf][i] = -dist * inv_sqrt_dh;
            }
        }

        // online softmax per row (16-lane half reductions, wave32)
#pragma unroll
        for (int i = 0; i < 8; ++i) {
            float rm = fmaxf(fmaxf(sc[0][i], sc[1][i]), fmaxf(sc[2][i], sc[3][i]));
#pragma unroll
            for (int off = 8; off >= 1; off >>= 1)
                rm = fmaxf(rm, __shfl_xor(rm, off, 16));
            float mnew  = fmaxf(m8[i], rm);
            float scale = __expf(m8[i] - mnew);
            m8[i] = mnew;
            float rs = 0.f;
#pragma unroll
            for (int nf = 0; nf < 4; ++nf) {
                float p = __expf(sc[nf][i] - mnew);
                sc[nf][i] = p;
                rs += p;
            }
#pragma unroll
            for (int off = 8; off >= 1; off >>= 1)
                rs += __shfl_xor(rs, off, 16);
            l8[i] = l8[i] * scale + rs;
            ofr[0][i] *= scale;
            ofr[1][i] *= scale;
        }

        // stage P through per-wave LDS to re-shape C-layout -> A-layout
#pragma unroll
        for (int nf = 0; nf < 4; ++nf)
#pragma unroll
            for (int i = 0; i < 8; ++i)
                Pb[wave][(half * 8 + i) * 68 + nf * 16 + r] = (_Float16)sc[nf][i];
        asm volatile("s_wait_dscnt 0" ::: "memory");  // same-wave LDS RAW

        // O += P @ V (two K=32 chunks, two dh halves)
#pragma unroll
        for (int kc = 0; kc < 2; ++kc) {
            v16h pa;
#pragma unroll
            for (int j = 0; j < 8; ++j) {
                int kb = kc * 32 + kmapA(j, half);
                const _Float16* pp = &Pb[wave][r * 68 + kb];
                pa[2 * j] = pp[0]; pa[2 * j + 1] = pp[1];
            }
#pragma unroll
            for (int nd = 0; nd < 2; ++nd) {
                v16h vf;
#pragma unroll
                for (int j = 0; j < 8; ++j) {
                    int kb = kc * 32 + kmapB(j, half);   // over keys
                    vf[2 * j]     = Vt[(kb)*34 + nd * 16 + r];
                    vf[2 * j + 1] = Vt[(kb + 1) * 34 + nd * 16 + r];
                }
                ofr[nd] = __builtin_amdgcn_wmma_f32_16x16x32_f16(
                    false, pa, false, vf, (short)0, ofr[nd], false, false);
            }
        }
        __syncthreads();
    }

    // finalize: O / l, store heads-merged [B,S,D] f16 for the Wo GEMM
#pragma unroll
    for (int nd = 0; nd < 2; ++nd)
#pragma unroll
        for (int i = 0; i < 8; ++i) {
            float o  = ofr[nd][i] / l8[i];
            int   s_ = q0 + half * 8 + i;
            AO[((size_t)b * SEQ + s_) * D_MODEL + h * DH + nd * 16 + r] = (_Float16)o;
        }
}

// ---------------------------------------------------------------------------
extern "C" void kernel_launch(void* const* d_in, const int* in_sizes, int n_in,
                              void* d_out, int out_size, void* d_ws, size_t ws_size,
                              hipStream_t stream)
{
    (void)in_sizes; (void)n_in; (void)out_size; (void)ws_size;
    const float* x   = (const float*)d_in[0];
    const float* Wq  = (const float*)d_in[1];
    const float* bq  = (const float*)d_in[2];
    const float* Wk  = (const float*)d_in[3];
    const float* bk  = (const float*)d_in[4];
    const float* Wv  = (const float*)d_in[5];
    const float* bv  = (const float*)d_in[6];
    const float* Wo  = (const float*)d_in[7];
    const float* bo  = (const float*)d_in[8];
    const float* Wfc = (const float*)d_in[9];
    const float* bfc = (const float*)d_in[10];

    const size_t BHS = (size_t)BATCH * NUM_HEADS * SEQ;
    const size_t BSD = (size_t)BATCH * SEQ * D_MODEL;

    char* ws = (char*)d_ws;
    size_t off = 0;
    auto alloc = [&](size_t bytes) {
        void* p = ws + off;
        off = (off + bytes + 255) & ~(size_t)255;
        return p;
    };
    _Float16* qh = (_Float16*)alloc(BHS * DH * sizeof(_Float16));
    _Float16* kh = (_Float16*)alloc(BHS * DH * sizeof(_Float16));
    _Float16* vh = (_Float16*)alloc(BHS * DH * sizeof(_Float16));
    float*    qq = (float*)alloc(BHS * sizeof(float));
    float*    kk = (float*)alloc(BHS * sizeof(float));
    _Float16* ao = (_Float16*)alloc(BSD * sizeof(_Float16));
    _Float16* y1 = (_Float16*)alloc(BSD * sizeof(_Float16));

    const int M = BATCH * SEQ;           // 4096
    dim3 gblk(256), ggrid(M / 64);

    gemm_wt_kernel<float><<<ggrid, gblk, 0, stream>>>(x, Wq, bq, qh, qq, M, 256, MODE_EXPMAP_HEADS);
    gemm_wt_kernel<float><<<ggrid, gblk, 0, stream>>>(x, Wk, bk, kh, kk, M, 256, MODE_EXPMAP_HEADS);
    gemm_wt_kernel<float><<<ggrid, gblk, 0, stream>>>(x, Wv, bv, vh, nullptr, M, 256, MODE_HEADS_F16);

    hyp_attn_kernel<<<dim3(SEQ / 64, NUM_HEADS, BATCH), dim3(128), 0, stream>>>(
        qh, kh, vh, qq, kk, ao);

    gemm_wt_kernel<_Float16><<<ggrid, gblk, 0, stream>>>(ao, Wo, bo, y1, nullptr, M, 256, MODE_F16_OUT);
    gemm_wt_kernel<_Float16><<<ggrid, gblk, 0, stream>>>(y1, Wfc, bfc, (float*)d_out, nullptr, M, 256, MODE_F32_OUT);
}
